// Informer_64888365908142
// MI455X (gfx1250) — matrix operand order repaired and checked by reading, not compiled, not run on hardware
//
#include <hip/hip_runtime.h>
#include <math.h>

// ---------------------------------------------------------------------------
// Informer forward for gfx1250 (MI455X).
// GEMMs: f16 operands (pre-converted; weights pre-transposed to [N,K]),
// staged into LDS with GLOBAL_LOAD_ASYNC_TO_LDS_B128 (ASYNCcnt), consumed by
// 2x2 V_WMMA_F32_16X16X32_F16 per wave (128x64 block tile, 8 waves), f32 accum.
// ProbSparse attention follows the reference structure with the deterministic
// (full-QK) sparsity measure.
// ---------------------------------------------------------------------------

typedef __attribute__((ext_vector_type(16))) _Float16 v16h;
typedef __attribute__((ext_vector_type(8)))  _Float16 v8h;
typedef __attribute__((ext_vector_type(8)))  float    v8f;

#define NB    8
#define DM    512
#define NH    8
#define DH    64
#define NTOP  35

// ------------------------- f32 -> f16 conversion ---------------------------
__global__ void informer_cvt(const float* __restrict__ src, _Float16* __restrict__ dst, int n)
{
    int i = blockIdx.x * blockDim.x + threadIdx.x;
    if (i < n) dst[i] = (_Float16)src[i];
}

// transpose + convert: wt[n*K + k] = (f16) w[k*N + n]
__global__ void informer_cvtT(const float* __restrict__ w, _Float16* __restrict__ wt,
                              int K, int N)
{
    int i = blockIdx.x * blockDim.x + threadIdx.x;
    if (i >= K * N) return;
    int k = i % K;
    int n = i / K;
    wt[i] = (_Float16)w[(size_t)k * N + n];
}

// ------------------------------ WMMA GEMM ----------------------------------
// C[M,N] = act((Ah[M,K] @ Wt[N,K]^T + bias) * scale*scale_c + shift)
// act: 0 none, 1 exact GELU, 2 ELU.
// 256 threads = 8 waves (4x2). Block tile 128(M) x 64(N); wave tile 32x32 =
// 2x2 WMMA fragments. Tiles staged in LDS via async loads.
__global__ __launch_bounds__(256)
void informer_gemm(const _Float16* __restrict__ Ah, const _Float16* __restrict__ Wt,
                   const float* __restrict__ bias, const float* __restrict__ scale,
                   const float* __restrict__ shift, float scale_c,
                   float* __restrict__ C, int M, int K, int N, int act)
{
    __shared__ _Float16 __attribute__((aligned(32))) sA[128 * 32];
    __shared__ _Float16 __attribute__((aligned(32))) sB[64 * 32];

    const int tid  = threadIdx.x;
    const int wave = tid >> 5;
    const int lane = tid & 31;
    const int ml   = lane & 15;
    const int hi   = lane >> 4;
    const int wm   = wave >> 1;            // 0..3  -> M sub-tile * 32
    const int wn   = wave & 1;             // 0..1  -> N sub-tile * 32
    const int m0g  = blockIdx.x * 128;
    const int n0g  = blockIdx.y * 64;

    // async-load thread mappings (clamped; stores are guarded later)
    int ar = tid >> 1;                     // A row 0..127
    const int ah = tid & 1;                // A k-half (16 f16 each)
    int arow = m0g + ar; if (arow >= M) arow = M - 1;
    int bn = tid >> 2;                     // B row (n) 0..63
    const int bq = tid & 3;                // B k-quarter (8 f16 each)
    int brow = n0g + bn; if (brow >= N) brow = N - 1;

    const unsigned ldsA = (unsigned)(size_t)&sA[ar * 32 + ah * 16];
    const unsigned ldsB = (unsigned)(size_t)&sB[bn * 32 + bq * 8];

    v8f acc00 = {0,0,0,0,0,0,0,0}, acc01 = acc00, acc10 = acc00, acc11 = acc00;

    for (int k0 = 0; k0 < K; k0 += 32) {
        __syncthreads();   // previous tile fully consumed
        {
            const _Float16* ga = Ah + (size_t)arow * K + k0 + ah * 16;
            const _Float16* gb = Wt + (size_t)brow * K + k0 + bq * 8;
            unsigned long long a0 = (unsigned long long)(size_t)ga;
            unsigned long long a1 = (unsigned long long)(size_t)(ga + 8);
            unsigned long long b0 = (unsigned long long)(size_t)gb;
            asm volatile("global_load_async_to_lds_b128 %0, %1, off"
                         :: "v"(ldsA), "v"(a0) : "memory");
            asm volatile("global_load_async_to_lds_b128 %0, %1, off"
                         :: "v"(ldsA + 16u), "v"(a1) : "memory");
            asm volatile("global_load_async_to_lds_b128 %0, %1, off"
                         :: "v"(ldsB), "v"(b0) : "memory");
            asm volatile("s_wait_asynccnt 0x0" ::: "memory");
        }
        __syncthreads();   // whole tile visible

        // A fragments (16x32): lane row = wm*32 + mt*16 + ml;
        // elems 0..7 -> k=hi*8+0..7, elems 8..15 -> k=hi*8+16..23
        const int ra0 = (wm * 32 + ml) * 32 + hi * 8;
        const int ra1 = (wm * 32 + 16 + ml) * 32 + hi * 8;
        v8h a0lo = *(const v8h*)&sA[ra0];
        v8h a0hi = *(const v8h*)&sA[ra0 + 16];
        v8h a1lo = *(const v8h*)&sA[ra1];
        v8h a1hi = *(const v8h*)&sA[ra1 + 16];
        v16h af0 = __builtin_shufflevector(a0lo, a0hi, 0,1,2,3,4,5,6,7,8,9,10,11,12,13,14,15);
        v16h af1 = __builtin_shufflevector(a1lo, a1hi, 0,1,2,3,4,5,6,7,8,9,10,11,12,13,14,15);

        // B fragments (32x16): lane col = wn*32 + nt*16 + ml;
        // elems 0..7 -> k=hi*8+0..7, elems 8..15 -> k=hi*8+16..23
        const int rb0 = (wn * 32 + ml) * 32 + hi * 8;
        const int rb1 = (wn * 32 + 16 + ml) * 32 + hi * 8;
        v8h b0lo = *(const v8h*)&sB[rb0];
        v8h b0hi = *(const v8h*)&sB[rb0 + 16];
        v8h b1lo = *(const v8h*)&sB[rb1];
        v8h b1hi = *(const v8h*)&sB[rb1 + 16];
        v16h bf0 = __builtin_shufflevector(b0lo, b0hi, 0,1,2,3,4,5,6,7,8,9,10,11,12,13,14,15);
        v16h bf1 = __builtin_shufflevector(b1lo, b1hi, 0,1,2,3,4,5,6,7,8,9,10,11,12,13,14,15);

        acc00 = __builtin_amdgcn_wmma_f32_16x16x32_f16(false, af0, false, bf0, (short)0, acc00, false, false);
        acc01 = __builtin_amdgcn_wmma_f32_16x16x32_f16(false, af0, false, bf1, (short)0, acc01, false, false);
        acc10 = __builtin_amdgcn_wmma_f32_16x16x32_f16(false, af1, false, bf0, (short)0, acc10, false, false);
        acc11 = __builtin_amdgcn_wmma_f32_16x16x32_f16(false, af1, false, bf1, (short)0, acc11, false, false);
    }

    // epilogue
    #pragma unroll
    for (int mt = 0; mt < 2; ++mt) {
        #pragma unroll
        for (int nt = 0; nt < 2; ++nt) {
            v8f acc = (mt == 0) ? (nt == 0 ? acc00 : acc01) : (nt == 0 ? acc10 : acc11);
            const int n = n0g + wn * 32 + nt * 16 + ml;
            if (n >= N) continue;
            const float bs  = scale ? scale[n] * scale_c : 1.0f;
            const float bb  = shift ? shift[n] : 0.0f;
            const float bv0 = bias ? bias[n] : 0.0f;
            #pragma unroll
            for (int r = 0; r < 8; ++r) {
                const int m = m0g + wm * 32 + mt * 16 + hi * 8 + r;
                if (m < M) {
                    float t = acc[r] + bv0;
                    t = t * bs + bb;
                    if (act == 1)      t = 0.5f * t * (1.0f + erff(t * 0.70710678118f));
                    else if (act == 2) t = t > 0.0f ? t : (expf(t) - 1.0f);
                    C[(size_t)m * N + n] = t;
                }
            }
        }
    }
}

// ---------------------- circular im2col (3-tap conv) -----------------------
__global__ void informer_im2col3(const float* __restrict__ x, float* __restrict__ out,
                                 int L, int C)
{
    int idx = blockIdx.x * blockDim.x + threadIdx.x;
    int total = NB * L * 3 * C;
    if (idx >= total) return;
    int c = idx % C;
    int t = (idx / C) % 3;
    int l = (idx / (3 * C)) % L;
    int b = idx / (3 * C * L);
    int ls = l + t - 1;
    if (ls < 0) ls += L;
    if (ls >= L) ls -= L;
    out[idx] = x[((size_t)b * L + ls) * C + c];
}

// ---------------------- positional + time-mark embedding -------------------
__global__ void informer_pos_time(float* __restrict__ y, const float* __restrict__ xmark,
                                  const float* __restrict__ time_w,
                                  const float* __restrict__ time_b, int L)
{
    int idx = blockIdx.x * blockDim.x + threadIdx.x;
    int total = NB * L * DM;
    if (idx >= total) return;
    int d = idx & (DM - 1);
    int l = (idx / DM) % L;
    int b = idx / (DM * L);
    int i2 = d & ~1;
    float freq = expf(-(float)i2 * (logf(10000.0f) / (float)DM));
    float ang = (float)l * freq;
    float pe = (d & 1) ? cosf(ang) : sinf(ang);
    float tm = time_b[d];
    #pragma unroll
    for (int m = 0; m < 4; ++m)
        tm += xmark[((size_t)b * L + l) * 4 + m] * time_w[m * DM + d];
    y[idx] += pe + tm;
}

// ---------------------- LayerNorm(x + a) * g + b ---------------------------
__global__ __launch_bounds__(256)
void informer_lnres(const float* __restrict__ x, const float* __restrict__ a,
                    const float* __restrict__ g, const float* __restrict__ bta,
                    float* __restrict__ dst)
{
    __shared__ float red[256];
    const int row = blockIdx.x;
    const int tid = threadIdx.x;
    const float* xr = x + (size_t)row * DM;
    float v0 = xr[tid], v1 = xr[tid + 256];
    if (a) { v0 += a[(size_t)row * DM + tid]; v1 += a[(size_t)row * DM + tid + 256]; }

    red[tid] = v0 + v1; __syncthreads();
    for (int o = 128; o > 0; o >>= 1) { if (tid < o) red[tid] += red[tid + o]; __syncthreads(); }
    const float mean = red[0] / (float)DM;
    __syncthreads();
    const float d0 = v0 - mean, d1 = v1 - mean;
    red[tid] = d0 * d0 + d1 * d1; __syncthreads();
    for (int o = 128; o > 0; o >>= 1) { if (tid < o) red[tid] += red[tid + o]; __syncthreads(); }
    const float rstd = rsqrtf(red[0] / (float)DM + 1e-5f);
    dst[(size_t)row * DM + tid]       = d0 * rstd * g[tid]       + bta[tid];
    dst[(size_t)row * DM + tid + 256] = d1 * rstd * g[tid + 256] + bta[tid + 256];
}

// ---------------------- ProbSparse measure M -------------------------------
__global__ void informer_mscore(const float* __restrict__ Q, const float* __restrict__ K,
                                float* __restrict__ Ms, int L)
{
    int idx = blockIdx.x * blockDim.x + threadIdx.x;
    if (idx >= NB * NH * L) return;
    int l = idx % L;
    int bh = idx / L;
    int h = bh & (NH - 1);
    int b = bh / NH;
    const float* q = Q + (((size_t)b * L + l) * NH + h) * DH;
    float qv[DH];
    #pragma unroll
    for (int d = 0; d < DH; ++d) qv[d] = q[d];
    float mx = -INFINITY, sm = 0.0f;
    for (int s = 0; s < L; ++s) {
        const float* kp = K + (((size_t)b * L + s) * NH + h) * DH;
        float dot = 0.0f;
        #pragma unroll 8
        for (int d = 0; d < DH; ++d) dot += qv[d] * kp[d];
        mx = fmaxf(mx, dot);
        sm += dot;
    }
    Ms[idx] = mx - sm / (float)L;
}

// ---------------------- iterative top-35 per (b,h) -------------------------
__global__ __launch_bounds__(256)
void informer_topk(const float* __restrict__ Ms, int* __restrict__ topi, int L)
{
    __shared__ float vals[1024];
    __shared__ float rv[256];
    __shared__ int   ri[256];
    const int bh = blockIdx.x;
    const int tid = threadIdx.x;
    for (int i = tid; i < L; i += 256) vals[i] = Ms[(size_t)bh * L + i];
    __syncthreads();
    for (int t = 0; t < NTOP; ++t) {
        float best = -INFINITY; int bi = 0;
        for (int i = tid; i < L; i += 256)
            if (vals[i] > best) { best = vals[i]; bi = i; }
        rv[tid] = best; ri[tid] = bi; __syncthreads();
        for (int o = 128; o > 0; o >>= 1) {
            if (tid < o && rv[tid + o] > rv[tid]) { rv[tid] = rv[tid + o]; ri[tid] = ri[tid + o]; }
            __syncthreads();
        }
        if (tid == 0) { topi[bh * NTOP + t] = ri[0]; vals[ri[0]] = -INFINITY; }
        __syncthreads();
    }
}

// ---------------------- mean(V) over keys ----------------------------------
__global__ void informer_meanv(const float* __restrict__ V, float* __restrict__ mv, int Lk)
{
    int idx = blockIdx.x * blockDim.x + threadIdx.x;
    if (idx >= NB * NH * DH) return;
    int d = idx & (DH - 1);
    int h = (idx >> 6) & (NH - 1);
    int b = idx >> 9;
    float s = 0.0f;
    for (int l = 0; l < Lk; ++l)
        s += V[(((size_t)b * Lk + l) * NH + h) * DH + d];
    mv[idx] = s / (float)Lk;
}

// ctx[b,l,h,d] = meanV[b,h,d]
__global__ void informer_bcast(const float* __restrict__ mv, float* __restrict__ ctx, int L)
{
    int idx = blockIdx.x * blockDim.x + threadIdx.x;
    if (idx >= NB * L * NH * DH) return;
    int d = idx & (DH - 1);
    int h = (idx >> 6) & (NH - 1);
    int b = idx / (L * NH * DH);
    ctx[idx] = mv[((size_t)b * NH + h) * DH + d];
}

// causal: ctx[b,h,l,d] = sum_{s<=l} V[b,s,h,d]   (ctx laid out [B,H,L,D])
__global__ void informer_cumsum(const float* __restrict__ V, float* __restrict__ ctx, int L)
{
    int idx = blockIdx.x * blockDim.x + threadIdx.x;
    if (idx >= NB * NH * DH) return;
    int d = idx & (DH - 1);
    int h = (idx >> 6) & (NH - 1);
    int b = idx >> 9;
    float acc = 0.0f;
    for (int l = 0; l < L; ++l) {
        acc += V[(((size_t)b * L + l) * NH + h) * DH + d];
        ctx[(((size_t)b * NH + h) * L + l) * DH + d] = acc;
    }
}

// ---------------------- per-row softmax attention --------------------------
// mode 0: enc top-row update, non-causal, ctx [B,L,H,D], row = topi[]
// mode 1: dec self top-row update, causal (keys <= row), ctx [B,H,L,D]
// mode 2: full cross attention, ctx [B,L,H,D], row = u
__global__ __launch_bounds__(256)
void informer_attnrow(const float* __restrict__ Q, const float* __restrict__ K,
                      const float* __restrict__ V, const int* __restrict__ topi,
                      float* __restrict__ ctx, int Lq, int Lk, int nU, int mode)
{
    __shared__ float sq[DH];
    __shared__ float sc[1024];
    __shared__ float red[256];
    __shared__ float part[256];

    const int task = blockIdx.x;
    const int u = task % nU;
    const int h = (task / nU) & (NH - 1);
    const int b = task / (nU * NH);
    const int tid = threadIdx.x;

    const int qrow = (mode == 2) ? u : topi[((size_t)b * NH + h) * NTOP + u];
    const int LkE  = (mode == 1) ? (qrow + 1) : Lk;

    if (tid < DH) sq[tid] = Q[(((size_t)b * Lq + qrow) * NH + h) * DH + tid];
    __syncthreads();

    for (int s = tid; s < LkE; s += 256) {
        const float* kp = K + (((size_t)b * Lk + s) * NH + h) * DH;
        float dot = 0.0f;
        #pragma unroll 8
        for (int d = 0; d < DH; ++d) dot += sq[d] * kp[d];
        sc[s] = dot * 0.125f;                    // 1/sqrt(64)
    }
    __syncthreads();

    float mx = -INFINITY;
    for (int s = tid; s < LkE; s += 256) mx = fmaxf(mx, sc[s]);
    red[tid] = mx; __syncthreads();
    for (int o = 128; o > 0; o >>= 1) { if (tid < o) red[tid] = fmaxf(red[tid], red[tid + o]); __syncthreads(); }
    mx = red[0]; __syncthreads();

    float se = 0.0f;
    for (int s = tid; s < LkE; s += 256) { float p = expf(sc[s] - mx); sc[s] = p; se += p; }
    red[tid] = se; __syncthreads();
    for (int o = 128; o > 0; o >>= 1) { if (tid < o) red[tid] += red[tid + o]; __syncthreads(); }
    se = red[0]; __syncthreads();

    const int d = tid & (DH - 1);
    const int g = tid >> 6;
    float acc = 0.0f;
    for (int s = g; s < LkE; s += 4)
        acc += sc[s] * V[(((size_t)b * Lk + s) * NH + h) * DH + d];
    part[tid] = acc; __syncthreads();
    if (g == 0) {
        float t = part[d] + part[64 + d] + part[128 + d] + part[192 + d];
        size_t off = (mode == 1)
            ? ((((size_t)b * NH + h) * Lq + qrow) * DH + d)
            : ((((size_t)b * Lq + qrow) * NH + h) * DH + d);
        ctx[off] = t / se;
    }
}

// ---------------------- maxpool k=3 s=2 pad=1 along L ----------------------
__global__ void informer_maxpool(const float* __restrict__ x, float* __restrict__ y, int L)
{
    const int Lo = L / 2;
    int idx = blockIdx.x * blockDim.x + threadIdx.x;
    if (idx >= NB * Lo * DM) return;
    int d = idx & (DM - 1);
    int j = (idx / DM) % Lo;
    int b = idx / (DM * Lo);
    int l0 = 2 * j - 1;
    float m = -INFINITY;
    #pragma unroll
    for (int t = 0; t < 3; ++t) {
        int l = l0 + t;
        if (l >= 0 && l < L) m = fmaxf(m, x[((size_t)b * L + l) * DM + d]);
    }
    y[idx] = m;
}

// ---------------------- final slice [:, -256:, :] --------------------------
__global__ void informer_copyout(const float* __restrict__ proj, float* __restrict__ out)
{
    int idx = blockIdx.x * blockDim.x + threadIdx.x;
    if (idx >= NB * 256 * 32) return;
    int c = idx & 31;
    int j = (idx >> 5) & 255;
    int b = idx >> 13;
    out[idx] = proj[(((size_t)b * 512) + 256 + j) * 32 + c];
}

// ===========================================================================
extern "C" void kernel_launch(void* const* d_in, const int* in_sizes, int n_in,
                              void* d_out, int out_size, void* d_ws, size_t ws_size,
                              hipStream_t stream)
{
    (void)in_sizes; (void)n_in; (void)out_size;
    #define PP(i) ((const float*)d_in[i])

    const float* x_enc      = PP(0);
    const float* x_mark_enc = PP(1);
    const float* x_dec      = PP(2);
    const float* x_mark_dec = PP(3);

    // params flattened as a JAX pytree (sorted dict keys recursively):
    // dec_emb 4..7 {conv_b,conv_w,time_b,time_w}
    // dec_layers[0] 8..33 {b1,b2, cross{bk,bo,bq,bv,wk,wo,wq,wv},
    //                      ln1_b,ln1_g,ln2_b,ln2_g,ln3_b,ln3_g, self{...}, w1,w2}
    // dec_norm_b=34, dec_norm_g=35
    // enc_convs[0]: b=36, bn_b=37, bn_g=38, w=39
    // enc_emb 40..43; enc_layers[l] base=44+16l:
    //   {bk,bo,bq,bv,wk,wo,wq,wv, b1,b2, ln1_b,ln1_g,ln2_b,ln2_g, w1,w2}
    // enc_norm_b=76, enc_norm_g=77, proj_b=78, proj_w=79

    // ---------------- workspace layout ------------------
    const size_t SZ_ACT = (size_t)NB * 1024 * DM;      // 4,194,304 f32
    const size_t SZ_BIG = (size_t)NB * 1024 * 2048;    // 16,777,216 f32
    float* wsf = (float*)d_ws;
    size_t off = 0;
    float* XE   = wsf + off; off += SZ_ACT;
    float* T0   = wsf + off; off += SZ_ACT;
    float* T1   = wsf + off; off += SZ_ACT;
    float* Qb   = wsf + off; off += SZ_ACT;
    float* Kb   = wsf + off; off += SZ_ACT;
    float* Vb   = wsf + off; off += SZ_ACT;
    float* CTX  = wsf + off; off += SZ_ACT;
    float* BIG  = wsf + off; off += SZ_BIG;
    float* XD   = wsf + off; off += (size_t)NB * 512 * DM;
    float* MS   = wsf + off; off += (size_t)NB * NH * 1024;
    float* MV   = wsf + off; off += (size_t)NB * NH * DH;
    float* PROJ = wsf + off; off += (size_t)NB * 512 * 32;
    int*   TOPI = (int*)(wsf + off); off += 4096;
    _Float16* AH = (_Float16*)(wsf + off); off += SZ_BIG / 2;      // M*K max (f16)
    _Float16* WT = (_Float16*)(wsf + off); off += (512 * 2048) / 2; // K*N max (f16)
    if (ws_size < off * sizeof(float)) return;

    const float bn_scale = 1.0f / sqrtf(1.0f + 1e-5f);

    auto gemm = [&](const float* A, const float* W, const float* bias,
                    const float* scale, const float* shift, float sc,
                    float* C, int M, int Kd, int N, int act) {
        informer_cvt<<<dim3((M * Kd + 255) / 256), 256, 0, stream>>>(A, AH, M * Kd);
        informer_cvtT<<<dim3((Kd * N + 255) / 256), 256, 0, stream>>>(W, WT, Kd, N);
        dim3 grid((M + 127) / 128, (N + 63) / 64);
        informer_gemm<<<grid, 256, 0, stream>>>(AH, WT, bias, scale, shift, sc, C, M, Kd, N, act);
    };
    auto elt = [&](int total) { return dim3((total + 255) / 256); };

    // ============================ ENCODER ==================================
    {
        int L = 1024, M = NB * L;
        informer_im2col3<<<elt(M * 3 * 32), 256, 0, stream>>>(x_enc, BIG, L, 32);
        gemm(BIG, PP(41), PP(40), nullptr, nullptr, 1.f, XE, M, 96, DM, 0);
        informer_pos_time<<<elt(M * DM), 256, 0, stream>>>(XE, x_mark_enc, PP(43), PP(42), L);

        int curL = L;
        for (int l = 0; l < 2; ++l) {
            const int base = 44 + 16 * l;
            const int Mr = NB * curL;
            gemm(XE, PP(base + 6), PP(base + 2), nullptr, nullptr, 1.f, Qb, Mr, DM, DM, 0);
            gemm(XE, PP(base + 4), PP(base + 0), nullptr, nullptr, 1.f, Kb, Mr, DM, DM, 0);
            gemm(XE, PP(base + 7), PP(base + 3), nullptr, nullptr, 1.f, Vb, Mr, DM, DM, 0);
            informer_mscore<<<elt(NB * NH * curL), 256, 0, stream>>>(Qb, Kb, MS, curL);
            informer_topk<<<NB * NH, 256, 0, stream>>>(MS, TOPI, curL);
            informer_meanv<<<elt(NB * NH * DH), 256, 0, stream>>>(Vb, MV, curL);
            informer_bcast<<<elt(Mr * DM), 256, 0, stream>>>(MV, CTX, curL);
            informer_attnrow<<<NB * NH * NTOP, 256, 0, stream>>>(Qb, Kb, Vb, TOPI, CTX,
                                                                 curL, curL, NTOP, 0);
            gemm(CTX, PP(base + 5), PP(base + 1), nullptr, nullptr, 1.f, T0, Mr, DM, DM, 0);
            informer_lnres<<<Mr, 256, 0, stream>>>(XE, T0, PP(base + 11), PP(base + 10), T1);
            gemm(T1, PP(base + 14), PP(base + 8), nullptr, nullptr, 1.f, BIG, Mr, DM, 2048, 1);
            gemm(BIG, PP(base + 15), PP(base + 9), nullptr, nullptr, 1.f, T0, Mr, 2048, DM, 0);
            informer_lnres<<<Mr, 256, 0, stream>>>(T1, T0, PP(base + 13), PP(base + 12), XE);
            if (l == 0) {
                informer_im2col3<<<elt(Mr * 3 * DM), 256, 0, stream>>>(XE, BIG, curL, DM);
                gemm(BIG, PP(39), PP(36), PP(38), PP(37), bn_scale, T0, Mr, 3 * DM, DM, 2);
                informer_maxpool<<<elt(NB * (curL / 2) * DM), 256, 0, stream>>>(T0, XE, curL);
                curL /= 2;
            }
        }
        informer_lnres<<<NB * curL, 256, 0, stream>>>(XE, nullptr, PP(77), PP(76), XE);
        // encoder output: XE, length 512
    }

    // ============================ DECODER ==================================
    {
        const int L = 512, M = NB * L, Lk = 512;
        informer_im2col3<<<elt(M * 3 * 32), 256, 0, stream>>>(x_dec, BIG, L, 32);
        gemm(BIG, PP(5), PP(4), nullptr, nullptr, 1.f, XD, M, 96, DM, 0);
        informer_pos_time<<<elt(M * DM), 256, 0, stream>>>(XD, x_mark_dec, PP(7), PP(6), L);

        // masked ProbSparse self-attention (mix=True)
        gemm(XD, PP(30), PP(26), nullptr, nullptr, 1.f, Qb, M, DM, DM, 0);
        gemm(XD, PP(28), PP(24), nullptr, nullptr, 1.f, Kb, M, DM, DM, 0);
        gemm(XD, PP(31), PP(27), nullptr, nullptr, 1.f, Vb, M, DM, DM, 0);
        informer_mscore<<<elt(NB * NH * L), 256, 0, stream>>>(Qb, Kb, MS, L);
        informer_topk<<<NB * NH, 256, 0, stream>>>(MS, TOPI, L);
        informer_cumsum<<<elt(NB * NH * DH), 256, 0, stream>>>(Vb, CTX, L);   // [B,H,L,D]
        informer_attnrow<<<NB * NH * NTOP, 256, 0, stream>>>(Qb, Kb, Vb, TOPI, CTX, L, L, NTOP, 1);
        gemm(CTX, PP(29), PP(25), nullptr, nullptr, 1.f, T0, M, DM, DM, 0);
        informer_lnres<<<M, 256, 0, stream>>>(XD, T0, PP(19), PP(18), T1);

        // full cross attention vs encoder output (XE)
        gemm(T1, PP(16), PP(12), nullptr, nullptr, 1.f, Qb, M, DM, DM, 0);
        gemm(XE, PP(14), PP(10), nullptr, nullptr, 1.f, Kb, NB * Lk, DM, DM, 0);
        gemm(XE, PP(17), PP(13), nullptr, nullptr, 1.f, Vb, NB * Lk, DM, DM, 0);
        informer_attnrow<<<NB * NH * L, 256, 0, stream>>>(Qb, Kb, Vb, nullptr, CTX, L, Lk, L, 2);
        gemm(CTX, PP(15), PP(11), nullptr, nullptr, 1.f, T0, M, DM, DM, 0);
        informer_lnres<<<M, 256, 0, stream>>>(T1, T0, PP(21), PP(20), XD);

        // FFN
        gemm(XD, PP(32), PP(8), nullptr, nullptr, 1.f, BIG, M, DM, 2048, 1);
        gemm(BIG, PP(33), PP(9), nullptr, nullptr, 1.f, T0, M, 2048, DM, 0);
        informer_lnres<<<M, 256, 0, stream>>>(XD, T0, PP(23), PP(22), T1);

        // final norm + projection + slice
        informer_lnres<<<M, 256, 0, stream>>>(T1, nullptr, PP(35), PP(34), T1);
        gemm(T1, PP(79), PP(78), nullptr, nullptr, 1.f, PROJ, M, DM, 32, 0);
        informer_copyout<<<elt(NB * 256 * 32), 256, 0, stream>>>(PROJ, (float*)d_out);
    }
    #undef PP
}
